// VectorQuantizer_78889959293401
// MI455X (gfx1250) — matrix operand order repaired and checked by reading, not compiled, not run on hardware
//
#include <hip/hip_runtime.h>
#include <hip/hip_bf16.h>

// ---------------------------------------------------------------------------
// VQ-VAE vector quantizer for MI455X (gfx1250, wave32, WMMA).
// Distance GEMM runs on v_wmma_f32_16x16x32_bf16 with a bf16 hi/lo split
// (3 WMMAs per k-step) for fp32-class argmin fidelity. Codebook chunks are
// staged with GLOBAL_LOAD_ASYNC_TO_LDS_B128 (ASYNCcnt) double-buffered.
// ---------------------------------------------------------------------------

#define K_CODES 1024
#define DIM     256
#define NVEC    65536          // 64*32*32
#define ZELEMS  16777216       // 64*256*32*32
#define SA      264            // padded LDS row stride (elements) -> no bank conflicts

typedef __attribute__((ext_vector_type(16))) __bf16 v16bf;
typedef __attribute__((ext_vector_type(8)))  float  v8f;

// ---- CDNA5 async global->LDS path (guarded; falls back to sync copy) ------
#if defined(__has_builtin)
# if __has_builtin(__builtin_amdgcn_global_load_async_to_lds_b128)
#  define HAVE_ASYNC 1
# endif
#endif
#ifndef HAVE_ASYNC
# define HAVE_ASYNC 0
#endif

#if HAVE_ASYNC
// Builtin signature (from clang diagnostic): first param is
// 'int __attribute__((vector_size(16))) __device__ *'. Build exact-typed
// pointers via integer round-trip (generic-LDS low 32 bits == LDS offset).
typedef int v4i_b128 __attribute__((vector_size(16)));
#define AS1P(p) ((__attribute__((address_space(1))) v4i_b128*)(unsigned long long)(const void*)(p))
#define AS3P(p) ((__attribute__((address_space(3))) v4i_b128*)(unsigned int)(unsigned long long)(const void*)(p))
# if __has_builtin(__builtin_amdgcn_s_wait_asynccnt)
#  define WAIT_ASYNC_8() __builtin_amdgcn_s_wait_asynccnt(8)
#  define WAIT_ASYNC_0() __builtin_amdgcn_s_wait_asynccnt(0)
# else
#  define WAIT_ASYNC_8() asm volatile("s_wait_asynccnt 0x8" ::: "memory")
#  define WAIT_ASYNC_0() asm volatile("s_wait_asynccnt 0x0" ::: "memory")
# endif
#else
#define WAIT_ASYNC_8() ((void)0)
#define WAIT_ASYNC_0() ((void)0)
#endif

__device__ __forceinline__ unsigned short f32_to_bf16_rne(float f) {
    union { float f; unsigned u; } c; c.f = f;
    unsigned u = c.u;
    unsigned r = u + 0x7FFFu + ((u >> 16) & 1u);
    return (unsigned short)(r >> 16);
}
__device__ __forceinline__ float bf16_to_f32(unsigned short h) {
    union { unsigned u; float f; } c; c.u = ((unsigned)h) << 16;
    return c.f;
}

// Issue one 16-code bf16 hi/lo chunk (global row-major) into padded LDS.
// Each wave moves rows [wave*4, wave*4+4): one async b128 per lane per row.
__device__ __forceinline__ void stage_b_chunk(
    const unsigned short* __restrict__ gh, const unsigned short* __restrict__ gl,
    unsigned short* lh, unsigned short* ll, int wave, int lane, int chunk) {
    #pragma unroll
    for (int i = 0; i < 4; i++) {
        int c  = wave * 4 + i;
        size_t ge = (size_t)(chunk * 16 + c) * DIM + lane * 8;
        int    le = c * SA + lane * 8;
#if HAVE_ASYNC
        __builtin_amdgcn_global_load_async_to_lds_b128(AS1P(gh + ge), AS3P(lh + le), 0, 0);
        __builtin_amdgcn_global_load_async_to_lds_b128(AS1P(gl + ge), AS3P(ll + le), 0, 0);
#else
        *reinterpret_cast<uint4*>(lh + le) = *reinterpret_cast<const uint4*>(gh + ge);
        *reinterpret_cast<uint4*>(ll + le) = *reinterpret_cast<const uint4*>(gl + ge);
#endif
    }
}

// ---------------------------------------------------------------------------
// Kernel 1: split codebook into bf16 hi/lo planes and compute ||c_k||^2.
// ---------------------------------------------------------------------------
__global__ __launch_bounds__(DIM) void prep_codebook(
    const float* __restrict__ cb,
    unsigned short* __restrict__ hi,
    unsigned short* __restrict__ lo,
    float* __restrict__ c2) {
    int k = blockIdx.x, d = threadIdx.x;
    float x = cb[k * DIM + d];
    unsigned short h = f32_to_bf16_rne(x);
    unsigned short l = f32_to_bf16_rne(x - bf16_to_f32(h));
    hi[k * DIM + d] = h;
    lo[k * DIM + d] = l;

    float s = x * x;
    #pragma unroll
    for (int o = 16; o >= 1; o >>= 1) s += __shfl_xor(s, o, 32);
    __shared__ float red[8];
    int lane = threadIdx.x & 31, w = threadIdx.x >> 5;
    if (lane == 0) red[w] = s;
    __syncthreads();
    if (threadIdx.x == 0) {
        float t = 0.f;
        #pragma unroll
        for (int i = 0; i < 8; i++) t += red[i];
        c2[k] = t;
    }
}

// ---------------------------------------------------------------------------
// Kernel 2: WMMA distance GEMM + running argmin.
// block = 128 threads (4 waves); each wave owns a 16-row M tile.
// grid = NVEC / 64 = 1024 blocks. Dynamic LDS: A hi/lo per wave + double-
// buffered 16-code B hi/lo chunks, staged via async-to-LDS.
// ---------------------------------------------------------------------------
__global__ __launch_bounds__(128) void argmin_wmma(
    const float* __restrict__ z_e,
    const unsigned short* __restrict__ cb_hi,
    const unsigned short* __restrict__ cb_lo,
    const float* __restrict__ c2,
    int* __restrict__ idx_out,
    float* __restrict__ idx_f_out) {
    extern __shared__ unsigned short smem[];
    unsigned short* Ahi = smem;                        // 4*16*SA
    unsigned short* Alo = Ahi + 4 * 16 * SA;           // 4*16*SA
    unsigned short* B0  = Alo + 4 * 16 * SA;           // [Bhi0 | Blo0]
    unsigned short* B1  = B0 + 2 * 16 * SA;            // [Bhi1 | Blo1]

    const int tid  = threadIdx.x;
    const int lane = tid & 31;
    const int wave = tid >> 5;
    const int r16  = lane & 15;   // row (A) / column n (B) / N (C)
    const int half = lane >> 4;   // lane half selects K sub-range per ISA layout

    const int base_n = blockIdx.x * 64 + wave * 16;

    unsigned short* myAhi = Ahi + wave * 16 * SA;
    unsigned short* myAlo = Alo + wave * 16 * SA;

    // Prologue: kick off async staging of B chunk 0 while we convert A.
    stage_b_chunk(cb_hi, cb_lo, B0, B0 + 16 * SA, wave, lane, 0);

    // --- Stage this wave's 16x256 A tile as bf16 hi/lo (fp32 split) ---
    {
        int n = base_n + r16;                  // tiles never cross the b-boundary
        int b = n >> 10, hw = n & 1023;
        const float* zbase = z_e + ((size_t)b * DIM) * 1024 + hw;
        for (int i = 0; i < 128; i++) {
            int d = i * 2 + half;
            float x = zbase[(size_t)d * 1024];
            unsigned short h = f32_to_bf16_rne(x);
            unsigned short l = f32_to_bf16_rne(x - bf16_to_f32(h));
            myAhi[r16 * SA + d] = h;
            myAlo[r16 * SA + d] = l;
        }
    }

    float minval[8];
    int   minidx[8];
    #pragma unroll
    for (int j = 0; j < 8; j++) { minval[j] = 3.4e38f; minidx[j] = 0; }

    for (int t = 0; t < K_CODES / 16; t++) {
        unsigned short* Bcur = (t & 1) ? B1 : B0;
        unsigned short* Bnxt = (t & 1) ? B0 : B1;
        // Prefetch chunk t+1 into the other buffer (last read at t-1,
        // released by the trailing barrier of that iteration).
        if (t + 1 < K_CODES / 16) {
            stage_b_chunk(cb_hi, cb_lo, Bnxt, Bnxt + 16 * SA, wave, lane, t + 1);
            // Async loads complete in order per wave: <=8 outstanding means
            // chunk t's 8 issues have landed in LDS.
            WAIT_ASYNC_8();
        } else {
            WAIT_ASYNC_0();
        }
        __syncthreads();   // publish chunk t across all waves

        const unsigned short* Bh = Bcur;
        const unsigned short* Bl = Bcur + 16 * SA;

        float myc2 = c2[t * 16 + r16];

        v8f acc = {0.f, 0.f, 0.f, 0.f, 0.f, 0.f, 0.f, 0.f};
        #pragma unroll
        for (int s = 0; s < 8; s++) {
            union { v16bf v; uint4 u[2]; } ah, al, bh, bl;
            {   // A frag: lane<16 -> K=[s32..+7]&[s32+16..+23]; lane>=16 -> +8
                int e0 = r16 * SA + s * 32 + half * 8;
                ah.u[0] = *reinterpret_cast<const uint4*>(myAhi + e0);
                ah.u[1] = *reinterpret_cast<const uint4*>(myAhi + e0 + 16);
                al.u[0] = *reinterpret_cast<const uint4*>(myAlo + e0);
                al.u[1] = *reinterpret_cast<const uint4*>(myAlo + e0 + 16);
            }
            {   // B frag: lane<16 -> K=[s32..+7]&[s32+16..+23]; lane>=16 -> +8
                int e0 = r16 * SA + s * 32 + half * 8;
                bh.u[0] = *reinterpret_cast<const uint4*>(Bh + e0);
                bh.u[1] = *reinterpret_cast<const uint4*>(Bh + e0 + 16);
                bl.u[0] = *reinterpret_cast<const uint4*>(Bl + e0);
                bl.u[1] = *reinterpret_cast<const uint4*>(Bl + e0 + 16);
            }
            // fp32-accurate dot via bf16x3: hi*hi + hi*lo + lo*hi
            acc = __builtin_amdgcn_wmma_f32_16x16x32_bf16(
                false, ah.v, false, bh.v, (short)0, acc, false, false);
            acc = __builtin_amdgcn_wmma_f32_16x16x32_bf16(
                false, ah.v, false, bl.v, (short)0, acc, false, false);
            acc = __builtin_amdgcn_wmma_f32_16x16x32_bf16(
                false, al.v, false, bh.v, (short)0, acc, false, false);
        }

        int code = t * 16 + r16;
        #pragma unroll
        for (int j = 0; j < 8; j++) {
            float d2 = myc2 - 2.0f * acc[j];   // ||x||^2 constant -> dropped
            if (d2 < minval[j]) { minval[j] = d2; minidx[j] = code; }
        }
        __syncthreads();   // all waves done reading Bcur before it is re-staged
    }

    // C layout: lane = N, VGPR slot j = row M=j (lanes 0-15) / M=j+8 (16-31).
    // Reduce min over the 16 lanes holding the same M (ties -> lowest index).
    #pragma unroll
    for (int j = 0; j < 8; j++) {
        float mv = minval[j]; int mi = minidx[j];
        #pragma unroll
        for (int o = 1; o < 16; o <<= 1) {
            float ov = __shfl_xor(mv, o, 16);
            int   oi = __shfl_xor(mi, o, 16);
            if (ov < mv || (ov == mv && oi < mi)) { mv = ov; mi = oi; }
        }
        if (r16 == 0) {
            int row = base_n + j + half * 8;
            idx_out[row]   = mi;
            idx_f_out[row] = (float)mi;
        }
    }
}

// ---------------------------------------------------------------------------
// Kernel 3: gather z_q_st (== z_q forward value), commit-loss partial sums,
// and scatter-add dw. One thread per NCHW element.
// ---------------------------------------------------------------------------
__global__ __launch_bounds__(256) void gather_scatter(
    const float* __restrict__ z_e,
    const float* __restrict__ cb,
    const int* __restrict__ idx,
    float* __restrict__ zq_out,
    float* __restrict__ dw,
    float* __restrict__ loss_acc) {
    size_t i = (size_t)blockIdx.x * blockDim.x + threadIdx.x;  // NCHW linear
    int hw = (int)(i & 1023);
    int d  = (int)((i >> 10) & 255);
    int b  = (int)(i >> 18);
    int n  = b * 1024 + hw;
    int k  = idx[n];
    float zq = cb[(size_t)k * DIM + d];
    float ze = z_e[i];
    zq_out[i] = zq;                       // straight-through forward == z_q
    atomicAdd(&dw[(size_t)k * DIM + d], ze);
    float diff = zq - ze;
    float s = diff * diff;
    #pragma unroll
    for (int o = 16; o >= 1; o >>= 1) s += __shfl_xor(s, o, 32);
    __shared__ float red[8];
    int lane = threadIdx.x & 31, w = threadIdx.x >> 5;
    if (lane == 0) red[w] = s;
    __syncthreads();
    if (threadIdx.x == 0) {
        float t = 0.f;
        #pragma unroll
        for (int q = 0; q < 8; q++) t += red[q];
        atomicAdd(loss_acc, t);
    }
}

// ---------------------------------------------------------------------------
// Kernel 4: bincount of indices.
// ---------------------------------------------------------------------------
__global__ __launch_bounds__(256) void count_kernel(
    const int* __restrict__ idx, float* __restrict__ counts) {
    int n = blockIdx.x * blockDim.x + threadIdx.x;
    atomicAdd(&counts[idx[n]], 1.0f);
}

// ---------------------------------------------------------------------------
// Kernel 5: EMA update, Laplace smoothing, dead-code reinit, final loss.
// ---------------------------------------------------------------------------
__global__ __launch_bounds__(256) void ema_finalize(
    const float* __restrict__ z_e,
    const float* __restrict__ ema_cs,
    const float* __restrict__ ema_dw,
    const float* __restrict__ counts,
    const float* __restrict__ dw,
    const float* __restrict__ loss_acc,
    float* __restrict__ out_loss,
    float* __restrict__ out_cb,
    float* __restrict__ out_cs,
    float* __restrict__ out_dw) {
    __shared__ float ncs_s[K_CODES];
    __shared__ float red[8];
    __shared__ float nsum_s;
    int tid = threadIdx.x;

    float part = 0.f;
    for (int k = tid; k < K_CODES; k += 256) {
        float v = ema_cs[k] * 0.99f + 0.01f * counts[k];
        ncs_s[k] = v;
        part += v;
    }
    #pragma unroll
    for (int o = 16; o >= 1; o >>= 1) part += __shfl_xor(part, o, 32);
    int lane = tid & 31, w = tid >> 5;
    if (lane == 0) red[w] = part;
    __syncthreads();
    if (tid == 0) {
        float t = 0.f;
        #pragma unroll
        for (int q = 0; q < 8; q++) t += red[q];
        nsum_s = t;
    }
    __syncthreads();
    float nsum = nsum_s;

    for (int k = tid; k < K_CODES; k += 256) {
        float ncs = ncs_s[k];
        bool dead = ncs < 1.0f;
        float smoothed = (ncs + 1e-5f) / (nsum + K_CODES * 1e-5f) * nsum;
        // NOTE: reference picks random rows via jax.random.randint(key(1));
        // we use a deterministic hash as the row selector (same-spirit reinit,
        // not bitwise-identical to JAX threefry).
        unsigned rr = (2654435761u * (unsigned)k) ^ 0x9E3779B9u;
        int rrow = (int)(rr & (NVEC - 1));
        int rb = rrow >> 10, rhw = rrow & 1023;
        for (int d = 0; d < DIM; d++) {
            float ndw = ema_dw[k * DIM + d] * 0.99f + 0.01f * dw[k * DIM + d];
            float ncb = ndw / smoothed;
            if (dead) {
                float rv = z_e[((size_t)rb * DIM + d) * 1024 + rhw];
                ncb = rv; ndw = rv;
            }
            out_cb[k * DIM + d] = ncb;
            out_dw[k * DIM + d] = ndw;
        }
        out_cs[k] = dead ? 1.0f : ncs;
    }
    if (tid == 0) out_loss[0] = 0.5f * loss_acc[0] / (float)ZELEMS;
}

// ---------------------------------------------------------------------------
extern "C" void kernel_launch(void* const* d_in, const int* in_sizes, int n_in,
                              void* d_out, int out_size, void* d_ws, size_t ws_size,
                              hipStream_t stream) {
    const float* z_e    = (const float*)d_in[0];
    const float* cb     = (const float*)d_in[1];
    const float* ema_cs = (const float*)d_in[2];
    const float* ema_dw = (const float*)d_in[3];
    float* out = (float*)d_out;

    // d_out layout (reference tuple order, flat):
    const size_t OFF_LOSS = (size_t)ZELEMS;               // 16777216
    const size_t OFF_IDX  = OFF_LOSS + 1;                 // 16777217
    const size_t OFF_CB   = OFF_IDX + NVEC;               // 16842753
    const size_t OFF_CS   = OFF_CB + (size_t)K_CODES*DIM; // 17104897
    const size_t OFF_DW   = OFF_CS + K_CODES;             // 17105921

    // Workspace layout (bytes):
    char* ws = (char*)d_ws;
    unsigned short* cb_hi = (unsigned short*)(ws);                         // 512KB
    unsigned short* cb_lo = (unsigned short*)(ws + 524288);                // 512KB
    float*          c2    = (float*)(ws + 1048576);                        // 4KB
    int*            idx   = (int*)(ws + 1052672);                          // 256KB
    float*          cnts  = (float*)(ws + 1314816);                        // 4KB (zeroed)
    float*          dw    = (float*)(ws + 1318912);                        // 1MB (zeroed)
    float*          lossa = (float*)(ws + 2367488);                        // 4B (zeroed)

    // Zero the accumulator region every call (graph-capture safe).
    (void)hipMemsetAsync(ws + 1314816, 0, 4096 + 1048576 + 4, stream);

    prep_codebook<<<K_CODES, DIM, 0, stream>>>(cb, cb_hi, cb_lo, c2);

    // LDS: A hi/lo (4 waves x 16 x SA each) + 2 double-buffered B hi/lo chunks
    const int lds_bytes = (2 * 4 * 16 * SA + 4 * 16 * SA) * 2;  // 101376
    argmin_wmma<<<NVEC / 64, 128, lds_bytes, stream>>>(
        z_e, cb_hi, cb_lo, c2, idx, out + OFF_IDX);

    gather_scatter<<<ZELEMS / 256, 256, 0, stream>>>(
        z_e, cb, idx, out /* z_q_st at offset 0 */, dw, lossa);

    count_kernel<<<NVEC / 256, 256, 0, stream>>>(idx, cnts);

    ema_finalize<<<1, 256, 0, stream>>>(
        z_e, ema_cs, ema_dw, cnts, dw, lossa,
        out + OFF_LOSS, out + OFF_CB, out + OFF_CS, out + OFF_DW);
}
